// VanillaRNN_47433618817627
// MI455X (gfx1250) — matrix operand order repaired and checked by reading, not compiled, not run on hardware
//
#include <hip/hip_runtime.h>
#include <math.h>

// ---------------------------------------------------------------------------
// VanillaRNN on MI455X (gfx1250): bf16 WMMA GEMM per timestep, fused tanh.
//   h_t = tanh(W_hh @ h_{t-1} + W_hx * x_t^T + b_h)      (512 serial steps)
//   y   = W_ph @ h_511 + b_p
// W_hh packed once per call into WMMA A-fragment bf16 layout; h lives in the
// WMMA B-fragment bf16 layout, ping-ponged between two ws buffers.
// Step kernel: single-wave workgroups (32 threads), 32x32 output tile per
// wave (2 A-frags x 2 B-frags -> 4 WMMAs per K-step), 512 blocks/step for
// maximal spread across WGPs.
// ---------------------------------------------------------------------------

typedef __attribute__((ext_vector_type(16))) __bf16 v16bf;
typedef __attribute__((ext_vector_type(8)))  __bf16 v8bf;
typedef __attribute__((ext_vector_type(8)))  float  v8f;

#define SEQ_LEN 512
#define HID     2048
#define BATCH   256
#define NCLS    10

#define KTILES  (HID / 32)     // 64  K-tiles of 32
#define MTILES  (HID / 16)     // 128 M-tiles of 16
#define NTILES  (BATCH / 16)   // 16  N-tiles of 16

__device__ __forceinline__ __bf16 f2bf(float f) {
    unsigned u = __builtin_bit_cast(unsigned, f);
    unsigned r = u + 0x7FFFu + ((u >> 16) & 1u);   // round-to-nearest-even
    unsigned short s = (unsigned short)(r >> 16);
    return __builtin_bit_cast(__bf16, s);
}
__device__ __forceinline__ float bf2f(__bf16 b) {
    unsigned short s = __builtin_bit_cast(unsigned short, b);
    unsigned u = ((unsigned)s) << 16;
    return __builtin_bit_cast(float, u);
}
// tanh(x) = 1 - 2/(e^{2x}+1); saturates to +/-1 for |x| large (exp->inf/0).
__device__ __forceinline__ float fast_tanh(float v) {
    float e = __expf(2.0f * v);
    return 1.0f - 2.0f / (e + 1.0f);
}

// ---------------------------------------------------------------------------
// Pack W_hh (fp32 row-major [2048][2048]) into bf16 WMMA A-fragments.
// Tile (mt, kt) stored as 32 lanes x 16 bf16 contiguous per lane.
//   lane <16 : M = 16*mt+lane,    elems = K k0+0..15
//   lane>=16 : M = 16*mt+lane-16, elems = K k0+16..31
// ---------------------------------------------------------------------------
__global__ __launch_bounds__(256) void pack_whh(const float* __restrict__ W,
                                                __bf16* __restrict__ Apack) {
    int gid = blockIdx.x * blockDim.x + threadIdx.x;
    if (gid >= MTILES * KTILES * 32) return;
    int lane  = gid & 31;
    int tile  = gid >> 5;                 // mt*KTILES + kt
    int kt    = tile & (KTILES - 1);
    int mt    = tile >> 6;                // log2(KTILES)=6
    int m     = mt * 16 + (lane & 15);
    int kbase = kt * 32 + ((lane >> 4) << 4);   // +16 for upper lane half
    const float* src = W + (size_t)m * HID;
    __bf16* dst = Apack + (size_t)tile * 512 + (size_t)lane * 16;
    v8bf lo, hi;
#pragma unroll
    for (int e = 0; e < 8; ++e) lo[e] = f2bf(src[kbase + e]);
#pragma unroll
    for (int e = 0; e < 8; ++e) hi[e] = f2bf(src[kbase + 8 + e]);
    *reinterpret_cast<v8bf*>(dst)     = lo;
    *reinterpret_cast<v8bf*>(dst + 8) = hi;
}

__global__ __launch_bounds__(256) void zero_u32(unsigned int* __restrict__ p, int n) {
    for (int i = blockIdx.x * blockDim.x + threadIdx.x; i < n; i += gridDim.x * blockDim.x)
        p[i] = 0u;
}

// ---------------------------------------------------------------------------
// One RNN timestep: Hnext = tanh(W_hh @ Hprev + W_hx x_t^T + b_h)
// Single wave per workgroup; wave tile = 32 (M) x 32 (N) = 4 WMMA C tiles.
// Grid = (2048/32, 256/32) = (64, 8) = 512 blocks.
// B-fragment layout, tile (kt, nt), 32 lanes x 16 bf16/lane:
//   lane <16 : N = 16*nt+lane,    elems = K k0+0..15
//   lane>=16 : N = 16*nt+lane-16, elems = K k0+16..31
// ---------------------------------------------------------------------------
__global__ __launch_bounds__(32)
void rnn_step(const __bf16* __restrict__ Apack,
              const __bf16* __restrict__ Bprev,
              __bf16* __restrict__ Bnext,
              const float* __restrict__ x,      // [BATCH][SEQ_LEN]
              const float* __restrict__ W_hx,   // [HID]
              const float* __restrict__ b_h,    // [HID]
              int t) {
    const int lane = threadIdx.x & 31;
    const int mt0  = blockIdx.x * 2;           // M-tiles mt0, mt0+1   [0,128)
    const int nt0  = blockIdx.y * 2;           // N-tiles nt0, nt0+1   [0,16)

    v8f acc[2][2] = {};

    const v16bf* Ab = reinterpret_cast<const v16bf*>(Apack)
                      + (size_t)mt0 * (KTILES * 32) + lane;
    const v16bf* Bb = reinterpret_cast<const v16bf*>(Bprev)
                      + (size_t)nt0 * 32 + lane;

#pragma unroll 2
    for (int kt = 0; kt < KTILES; ++kt) {
        v16bf a0 = Ab[(size_t)kt * 32];
        v16bf a1 = Ab[(size_t)kt * 32 + (size_t)KTILES * 32];
        v16bf b0 = Bb[(size_t)kt * (NTILES * 32)];
        v16bf b1 = Bb[(size_t)kt * (NTILES * 32) + 32];
        acc[0][0] = __builtin_amdgcn_wmma_f32_16x16x32_bf16(false, a0, false, b0,
                                                            (short)0, acc[0][0], false, false);
        acc[0][1] = __builtin_amdgcn_wmma_f32_16x16x32_bf16(false, a0, false, b1,
                                                            (short)0, acc[0][1], false, false);
        acc[1][0] = __builtin_amdgcn_wmma_f32_16x16x32_bf16(false, a1, false, b0,
                                                            (short)0, acc[1][0], false, false);
        acc[1][1] = __builtin_amdgcn_wmma_f32_16x16x32_bf16(false, a1, false, b1,
                                                            (short)0, acc[1][1], false, false);
    }

    // ---- fused epilogue: + outer(W_hx, x_t) + b_h, tanh, store B-fragments.
    // Each lane's 8 accumulator rows are 8 consecutive K values in the next-
    // step B-fragment layout -> one 16-byte store per C tile.
    const int half = lane >> 4;           // 0: rows r, 1: rows r+8 of a tile
    const int ln   = lane & 15;

#pragma unroll
    for (int mi = 0; mi < 2; ++mi) {
        const int mrow0 = (mt0 + mi) * 16 + half * 8;   // multiple of 8
        const int kk    = mrow0 & 31;                   // 0, 8, 16 or 24
        const size_t rowbase = ((size_t)(mrow0 >> 5) * NTILES) * 32;
        float wv[8], bv[8];
#pragma unroll
        for (int r = 0; r < 8; ++r) {
            wv[r] = W_hx[mrow0 + r];
            bv[r] = b_h[mrow0 + r];
        }
#pragma unroll
        for (int nj = 0; nj < 2; ++nj) {
            const int n = (nt0 + nj) * 16 + ln;
            const float xv = x[(size_t)n * SEQ_LEN + t];
            v8bf hv;
#pragma unroll
            for (int r = 0; r < 8; ++r)
                hv[r] = f2bf(fast_tanh(acc[mi][nj][r] + wv[r] * xv + bv[r]));
            size_t idx = (rowbase + (size_t)(nt0 + nj) * 32 + (ln + (kk & 16))) * 16
                         + (kk & 15);
            *reinterpret_cast<v8bf*>(Bnext + idx) = hv;   // 16B aligned
        }
    }
}

// ---------------------------------------------------------------------------
// y[c][n] = b_p[c] + sum_k W_ph[c][k] * h[k][n]   (10 x 256, reads packed bf16)
// ---------------------------------------------------------------------------
__global__ __launch_bounds__(128)
void proj(const __bf16* __restrict__ Hpack,
          const float* __restrict__ W_ph,   // [NCLS][HID]
          const float* __restrict__ b_p,    // [NCLS]
          float* __restrict__ out) {        // [NCLS][BATCH]
    int n = blockIdx.x * blockDim.x + threadIdx.x;
    if (n >= BATCH) return;
    const int nt = n >> 4;
    const int ln = n & 15;
    float acc[NCLS];
#pragma unroll
    for (int c = 0; c < NCLS; ++c) acc[c] = b_p[c];
    for (int k = 0; k < HID; ++k) {
        const int kt    = k >> 5;
        const int kk    = k & 31;
        const int lhalf = kk & 16;
        const int elem  = kk & 15;
        size_t idx = ((((size_t)kt * NTILES + nt) * 32) + (ln + lhalf)) * 16 + elem;
        float h = bf2f(Hpack[idx]);
#pragma unroll
        for (int c = 0; c < NCLS; ++c) acc[c] += W_ph[(size_t)c * HID + k] * h;
    }
#pragma unroll
    for (int c = 0; c < NCLS; ++c) out[(size_t)c * BATCH + n] = acc[c];
}

// ---------------------------------------------------------------------------
extern "C" void kernel_launch(void* const* d_in, const int* in_sizes, int n_in,
                              void* d_out, int out_size, void* d_ws, size_t ws_size,
                              hipStream_t stream) {
    const float* x    = (const float*)d_in[0];   // [256][512]
    const float* W_hx = (const float*)d_in[1];   // [2048][1]
    const float* W_hh = (const float*)d_in[2];   // [2048][2048]
    const float* W_ph = (const float*)d_in[3];   // [10][2048]
    const float* b_h  = (const float*)d_in[4];   // [2048]
    const float* b_p  = (const float*)d_in[5];   // [10]
    float* out = (float*)d_out;

    char* ws = (char*)d_ws;
    __bf16* Apack = (__bf16*)(ws);                          // 8 MB
    __bf16* H0    = (__bf16*)(ws + ((size_t)8 << 20));      // 1 MB
    __bf16* H1    = (__bf16*)(ws + ((size_t)9 << 20));      // 1 MB

    // 1) pack weights to WMMA A-fragments (bf16)
    pack_whh<<<(MTILES * KTILES * 32 + 255) / 256, 256, 0, stream>>>(W_hh, Apack);
    // 2) h_0 = 0
    zero_u32<<<256, 256, 0, stream>>>((unsigned int*)H0, (HID * BATCH * 2) / 4);
    // 3) 512 serial steps (grid-wide sync = kernel boundary); single-wave WGs
    for (int t = 0; t < SEQ_LEN; ++t) {
        const __bf16* cur = (t & 1) ? H1 : H0;
        __bf16*       nxt = (t & 1) ? H0 : H1;
        rnn_step<<<dim3(HID / 32, BATCH / 32), 32, 0, stream>>>(
            Apack, cur, nxt, x, W_hx, b_h, t);
    }
    // after t=511 (odd), last write went to H0
    // 4) output projection
    proj<<<(BATCH + 127) / 128, 128, 0, stream>>>(H0, W_ph, b_p, out);
}
